// GCAttention_75161927680451
// MI455X (gfx1250) — matrix operand order repaired and checked by reading, not compiled, not run on hardware
//
#include <hip/hip_runtime.h>
#include <hip/hip_bf16.h>

typedef __attribute__((ext_vector_type(16))) _Float16 v16h;
typedef __attribute__((ext_vector_type(8)))  float    v8f;

#define C_DIM   768
#define CR_DIM  192
#define B_DIM   8
#define N_DIM   1024
#define LN_EPSF 1e-5f

// ---------------- Phase 1: gc = mean_n x  (streaming pass #1 over x) ----------------

// 384 blocks x 256 thr: block = (b, n-slice of 64 rows, c-chunk of 256)
__global__ void k_partial_sum(const float* __restrict__ x, float* __restrict__ partial) {
    int blk = blockIdx.x;
    int ch  = blk % 3;
    int s   = (blk / 3) % 16;
    int b   = blk / 48;
    int c   = ch * 256 + threadIdx.x;
    const float* xp = x + ((size_t)b * N_DIM + (size_t)s * 64) * C_DIM + c;
    float acc = 0.f;
    #pragma unroll 8
    for (int r = 0; r < 64; ++r) acc += xp[(size_t)r * C_DIM];
    partial[(b * 16 + s) * C_DIM + c] = acc;
}

// finalize: gc16 = f16(mean), padded to 16 rows (rows 8..15 zero)
__global__ void k_finalize_gc(const float* __restrict__ partial, _Float16* __restrict__ gc16) {
    int idx = blockIdx.x * blockDim.x + threadIdx.x;   // < 16*768
    int r = idx / C_DIM, c = idx % C_DIM;
    if (r < B_DIM) {
        float acc = 0.f;
        #pragma unroll
        for (int s = 0; s < 16; ++s) acc += partial[(r * 16 + s) * C_DIM + c];
        gc16[idx] = (_Float16)(acc * (1.0f / N_DIM));
    } else {
        gc16[idx] = (_Float16)0.f;
    }
}

// ---------------- WMMA fragment helpers (contiguous k = 16h + i order, wave32) ----------------

// A 16x32 f16 (row-major source, leading dim lda):
//   lane<16: M=lane,   half i -> K = kb + i
//   lane>=16: M=lane-16, same + 16
__device__ __forceinline__ v16h load_a_frag(const _Float16* __restrict__ A, int lda,
                                            int kb, int lane) {
    int m  = lane & 15;
    int kh = (lane >> 4) << 4;           // 0 or 16
    v16h a;
    #pragma unroll
    for (int i = 0; i < 16; ++i) {
        int k = kb + kh + i;
        a[i] = A[m * lda + k];
    }
    return a;
}

// B 32x16 from fp32 row-major W (K x N, leading dim ldb), column tile n0:
//   half i -> K = kb + i + 16*(lane>=16), col = n0 + (lane&15)
__device__ __forceinline__ v16h load_b_frag_f32(const float* __restrict__ W, int ldb,
                                                int n0, int kb, int lane) {
    int n  = n0 + (lane & 15);
    int kh = (lane >> 4) << 4;           // 0 or 16
    v16h b;
    #pragma unroll
    for (int i = 0; i < 16; ++i)
        b[i] = (_Float16)W[(size_t)(kb + kh + i) * ldb + n];
    return b;
}

// ---------------- Phase 2: tiny per-batch GEMM chain via v_wmma_f32_16x16x32_f16 ----

// tiles 0..47:  vgc = gc @ Wv + bv   (16x768 fp32, padded rows harmless)
// tiles 48..59: h1  = relu(gc @ W1 + b1)  (16x192 f16)
__global__ void k_gemm_stage1(const _Float16* __restrict__ gc16,
                              const float* __restrict__ Wv, const float* __restrict__ bv,
                              const float* __restrict__ W1, const float* __restrict__ b1,
                              float* __restrict__ vgc32, _Float16* __restrict__ h1_16) {
    int lane = threadIdx.x;
    int tile = blockIdx.x;
    v8f acc = {};
    if (tile < 48) {
        int n0 = tile * 16;
        for (int kb = 0; kb < C_DIM; kb += 32) {
            v16h a = load_a_frag(gc16, C_DIM, kb, lane);
            v16h b = load_b_frag_f32(Wv, C_DIM, n0, kb, lane);
            acc = __builtin_amdgcn_wmma_f32_16x16x32_f16(false, a, false, b,
                                                         (short)0, acc, false, false);
        }
        int n  = n0 + (lane & 15);
        int mb = (lane >> 4) * 8;
        float bias = bv[n];
        #pragma unroll
        for (int r = 0; r < 8; ++r) vgc32[(mb + r) * C_DIM + n] = acc[r] + bias;
    } else {
        int n0 = (tile - 48) * 16;
        for (int kb = 0; kb < C_DIM; kb += 32) {
            v16h a = load_a_frag(gc16, C_DIM, kb, lane);
            v16h b = load_b_frag_f32(W1, CR_DIM, n0, kb, lane);
            acc = __builtin_amdgcn_wmma_f32_16x16x32_f16(false, a, false, b,
                                                         (short)0, acc, false, false);
        }
        int n  = n0 + (lane & 15);
        int mb = (lane >> 4) * 8;
        float bias = b1[n];
        #pragma unroll
        for (int r = 0; r < 8; ++r) {
            float h = acc[r] + bias;
            h1_16[(mb + r) * CR_DIM + n] = (_Float16)fmaxf(h, 0.f);
        }
    }
}

// ca = sigmoid(h1 @ W2 + b2);  u = ca * vgc  (stored f16 16x768)
__global__ void k_gemm_stage2(const _Float16* __restrict__ h1_16,
                              const float* __restrict__ W2, const float* __restrict__ b2,
                              const float* __restrict__ vgc32, _Float16* __restrict__ u16) {
    int lane = threadIdx.x;
    int n0 = blockIdx.x * 16;
    v8f acc = {};
    for (int kb = 0; kb < CR_DIM; kb += 32) {
        v16h a = load_a_frag(h1_16, CR_DIM, kb, lane);
        v16h b = load_b_frag_f32(W2, C_DIM, n0, kb, lane);
        acc = __builtin_amdgcn_wmma_f32_16x16x32_f16(false, a, false, b,
                                                     (short)0, acc, false, false);
    }
    int n  = n0 + (lane & 15);
    int mb = (lane >> 4) * 8;
    float bias = b2[n];
    #pragma unroll
    for (int r = 0; r < 8; ++r) {
        float ca = 1.f / (1.f + __expf(-(acc[r] + bias)));
        u16[(mb + r) * C_DIM + n] = (_Float16)(ca * vgc32[(mb + r) * C_DIM + n]);
    }
}

// t = u @ Wo + bo   (8x768 fp32, only real rows stored)
__global__ void k_gemm_stage3(const _Float16* __restrict__ u16,
                              const float* __restrict__ Wo, const float* __restrict__ bo,
                              float* __restrict__ t32) {
    int lane = threadIdx.x;
    int n0 = blockIdx.x * 16;
    v8f acc = {};
    for (int kb = 0; kb < C_DIM; kb += 32) {
        v16h a = load_a_frag(u16, C_DIM, kb, lane);
        v16h b = load_b_frag_f32(Wo, C_DIM, n0, kb, lane);
        acc = __builtin_amdgcn_wmma_f32_16x16x32_f16(false, a, false, b,
                                                     (short)0, acc, false, false);
    }
    if (lane < 16) {                       // rows 0..7 live in lanes 0..15
        int n = n0 + lane;
        float bias = bo[n];
        #pragma unroll
        for (int r = 0; r < 8; ++r) t32[r * C_DIM + n] = acc[r] + bias;
    }
}

// ---------------- Phase 3: y = LayerNorm(x + t[b]) (streaming pass #2, L2-resident x) ----

// 1024 blocks x 256 thr: one wave32 per row (8 rows/block), float4 IO, shuffle reduce.
__global__ void k_residual_ln(const float* __restrict__ x, const float* __restrict__ t32,
                              const float* __restrict__ gamma, const float* __restrict__ beta,
                              float* __restrict__ y) {
    int wave = threadIdx.x >> 5;
    int lane = threadIdx.x & 31;
    int row  = blockIdx.x * 8 + wave;           // 0 .. B*N-1
    int b    = row >> 10;
    const float4* xr = (const float4*)(x + (size_t)row * C_DIM);
    const float4* tr = (const float4*)(t32 + b * C_DIM);
    float4 v[6];
    float s1 = 0.f, s2 = 0.f;
    #pragma unroll
    for (int j = 0; j < 6; ++j) {
        float4 xv = xr[j * 32 + lane];
        float4 tv = tr[j * 32 + lane];
        float4 w = make_float4(xv.x + tv.x, xv.y + tv.y, xv.z + tv.z, xv.w + tv.w);
        v[j] = w;
        s1 += w.x + w.y + w.z + w.w;
        s2 += w.x * w.x + w.y * w.y + w.z * w.z + w.w * w.w;
    }
    #pragma unroll
    for (int off = 16; off >= 1; off >>= 1) {
        s1 += __shfl_xor(s1, off, 32);
        s2 += __shfl_xor(s2, off, 32);
    }
    float mean = s1 * (1.f / C_DIM);
    float var  = s2 * (1.f / C_DIM) - mean * mean;
    float rstd = rsqrtf(var + LN_EPSF);
    const float4* gr = (const float4*)gamma;
    const float4* br = (const float4*)beta;
    float4* yr = (float4*)(y + (size_t)row * C_DIM);
    #pragma unroll
    for (int j = 0; j < 6; ++j) {
        int idx = j * 32 + lane;
        float4 g = gr[idx], be = br[idx], w = v[j];
        yr[idx] = make_float4((w.x - mean) * rstd * g.x + be.x,
                              (w.y - mean) * rstd * g.y + be.y,
                              (w.z - mean) * rstd * g.z + be.z,
                              (w.w - mean) * rstd * g.w + be.w);
    }
}

// ---------------- Launch ----------------

extern "C" void kernel_launch(void* const* d_in, const int* in_sizes, int n_in,
                              void* d_out, int out_size, void* d_ws, size_t ws_size,
                              hipStream_t stream) {
    // setup_inputs order: x,Wq,bq,Wk,bk,Wv,bv,W1,b1,W2,b2,Wo,bo,gamma,beta
    const float* x     = (const float*)d_in[0];
    const float* Wv    = (const float*)d_in[5];
    const float* bv    = (const float*)d_in[6];
    const float* W1    = (const float*)d_in[7];
    const float* b1    = (const float*)d_in[8];
    const float* W2    = (const float*)d_in[9];
    const float* b2    = (const float*)d_in[10];
    const float* Wo    = (const float*)d_in[11];
    const float* bo    = (const float*)d_in[12];
    const float* gamma = (const float*)d_in[13];
    const float* beta  = (const float*)d_in[14];
    float* y = (float*)d_out;

    char* ws = (char*)d_ws;
    size_t off = 0;
    float*    partial = (float*)(ws + off);    off += (size_t)B_DIM * 16 * C_DIM * sizeof(float);
    _Float16* gc16    = (_Float16*)(ws + off); off += (size_t)16 * C_DIM * sizeof(_Float16);
    float*    vgc32   = (float*)(ws + off);    off += (size_t)16 * C_DIM * sizeof(float);
    _Float16* h1_16   = (_Float16*)(ws + off); off += (size_t)16 * CR_DIM * sizeof(_Float16);
    _Float16* u16     = (_Float16*)(ws + off); off += (size_t)16 * C_DIM * sizeof(_Float16);
    float*    t32     = (float*)(ws + off);    off += (size_t)B_DIM * C_DIM * sizeof(float);

    k_partial_sum <<<B_DIM * 16 * 3, 256, 0, stream>>>(x, partial);
    k_finalize_gc <<<(16 * C_DIM) / 256, 256, 0, stream>>>(partial, gc16);
    k_gemm_stage1 <<<60, 32, 0, stream>>>(gc16, Wv, bv, W1, b1, vgc32, h1_16);
    k_gemm_stage2 <<<48, 32, 0, stream>>>(h1_16, W2, b2, vgc32, u16);
    k_gemm_stage3 <<<48, 32, 0, stream>>>(u16, Wo, bo, t32);
    k_residual_ln <<<(B_DIM * N_DIM) / 8, 256, 0, stream>>>(x, t32, gamma, beta, y);
}
